// _PointnetSAModuleBase_15762529976895
// MI455X (gfx1250) — matrix operand order repaired and checked by reading, not compiled, not run on hardware
//
#include <hip/hip_runtime.h>
#include <hip/hip_bf16.h>

typedef _Float16 v8h  __attribute__((ext_vector_type(8)));
typedef _Float16 v16h __attribute__((ext_vector_type(16)));
typedef float    v8f  __attribute__((ext_vector_type(8)));

#define NBATCH   8
#define NPTS     8192
#define NPOINT   2048
#define NSAMPLE  32
#define CIN      64
#define C1       67          // 3 + CIN
#define H1DIM    128
#define H2DIM    256
#define RAD2     0.16f       // 0.4^2

#define G_PITCH  104         // halves per column for g   (16B aligned rows, conflict-free)
#define H_PITCH  136         // halves per column for h1  (16B aligned rows, conflict-free)
#define NCOL     128         // columns (4 query points * 32 samples) per workgroup
#define G_BYTES  (NCOL * G_PITCH * 2)   // 26624
#define H_BYTES  (NCOL * H_PITCH * 2)   // 34816

// ---------------------------------------------------------------------------
// K-index mapping for the 16-bit A operand of V_WMMA_F32_16X16X32_F16
// (lanes 0-15 M=0-15; element h of lane half hi holds K = 16*hi + h,
//  contiguous order matching the B operand, two packed halves per VGPR)
// ---------------------------------------------------------------------------
__device__ __host__ __forceinline__ int a_k_map(int kfrag, int lane, int h) {
    int hi = lane >> 4;
    return kfrag * 32 + hi * 16 + h;
}

// ---------------------------------------------------------------------------
// Pre-swizzle W1 [128x67] and W2 [256x128] into fragment-major f16 layout:
//   w_s[((mt*KF + kf)*32 + lane)*16 + h]  ->  one 32B contiguous load per lane
// ---------------------------------------------------------------------------
__global__ void prep_weights(const float* __restrict__ W1, const float* __restrict__ W2,
                             _Float16* __restrict__ w1s, _Float16* __restrict__ w2s) {
    int t = blockIdx.x * blockDim.x + threadIdx.x;
    const int N1 = 8 * 3 * 32 * 16;      // 12288
    const int N2 = 16 * 4 * 32 * 16;     // 32768
    if (t < N1) {
        int h = t & 15, lane = (t >> 4) & 31, r = t >> 9;
        int kf = r % 3, mt = r / 3;
        int m = mt * 16 + (lane & 15);
        int k = a_k_map(kf, lane, h);
        w1s[t] = (k < C1) ? (_Float16)W1[m * C1 + k] : (_Float16)0.f;
    } else if (t < N1 + N2) {
        int u = t - N1;
        int h = u & 15, lane = (u >> 4) & 31, r = u >> 9;
        int kf = r & 3, mt = r >> 2;
        int m = mt * 16 + (lane & 15);
        int k = a_k_map(kf, lane, h);
        w2s[u] = (_Float16)W2[m * H1DIM + k];
    }
}

// ---------------------------------------------------------------------------
// Tiled transpose: feat [B,64,8192] -> featT [B,8192,64] (coalesced both ways)
// makes the per-point gather read 256B contiguous instead of 64 strided dwords
// ---------------------------------------------------------------------------
__global__ void transpose_feat(const float* __restrict__ feat, float* __restrict__ featT) {
    __shared__ float tile[32][65];
    int blk   = blockIdx.x;            // NBATCH * 256 blocks
    int b     = blk >> 8;
    int pbase = (blk & 255) * 32;
    int tid   = threadIdx.x;
    for (int c = tid >> 5; c < 64; c += 8) {
        int p = tid & 31;
        tile[p][c] = feat[((size_t)b * 64 + c) * NPTS + pbase + p];
    }
    __syncthreads();
    for (int p = tid >> 6; p < 32; p += 4) {
        int c = tid & 63;
        featT[((size_t)b * NPTS + pbase + p) * 64 + c] = tile[p][c];
    }
}

// ---------------------------------------------------------------------------
// Furthest point sampling: one 1024-thread workgroup per batch.
// 8 points per thread kept in registers; two-level argmax reduction with
// first-index tie-break (matches jnp.argmax). Writes new_xyz directly.
// ---------------------------------------------------------------------------
__global__ __launch_bounds__(1024)
void fps_kernel(const float* __restrict__ xyz, float* __restrict__ newxyz) {
    const int b   = blockIdx.x;
    const int tid = threadIdx.x;
    const float* X = xyz + (size_t)b * NPTS * 3;

    float px[8], py[8], pz[8], mind[8];
#pragma unroll
    for (int i = 0; i < 8; ++i) {
        int p = i * 1024 + tid;
        px[i] = X[p * 3 + 0];
        py[i] = X[p * 3 + 1];
        pz[i] = X[p * 3 + 2];
        mind[i] = 1e10f;
    }

    __shared__ float selx, sely, selz;
    __shared__ float wval[32];
    __shared__ int   widx[32];
    if (tid == 0) { selx = X[0]; sely = X[1]; selz = X[2]; }
    __syncthreads();

    const int lane = tid & 31, wid = tid >> 5;

    for (int s = 0; s < NPOINT; ++s) {
        float sx = selx, sy = sely, sz = selz;
        if (tid == 0) {
            newxyz[((size_t)b * NPOINT + s) * 3 + 0] = sx;
            newxyz[((size_t)b * NPOINT + s) * 3 + 1] = sy;
            newxyz[((size_t)b * NPOINT + s) * 3 + 2] = sz;
        }
        float bv = -1.f; int bi = 0x7fffffff;
#pragma unroll
        for (int i = 0; i < 8; ++i) {
            float dx = px[i] - sx, dy = py[i] - sy, dz = pz[i] - sz;
            float d = dx * dx + dy * dy + dz * dz;
            float m = fminf(mind[i], d);
            mind[i] = m;
            if (m > bv) { bv = m; bi = i * 1024 + tid; }
        }
#pragma unroll
        for (int off = 16; off > 0; off >>= 1) {
            float ov = __shfl_down(bv, off);
            int   oi = __shfl_down(bi, off);
            if (ov > bv || (ov == bv && oi < bi)) { bv = ov; bi = oi; }
        }
        if (lane == 0) { wval[wid] = bv; widx[wid] = bi; }
        __syncthreads();
        if (wid == 0) {
            bv = wval[lane]; bi = widx[lane];
#pragma unroll
            for (int off = 16; off > 0; off >>= 1) {
                float ov = __shfl_down(bv, off);
                int   oi = __shfl_down(bi, off);
                if (ov > bv || (ov == bv && oi < bi)) { bv = ov; bi = oi; }
            }
            if (lane == 0) {
                selx = X[bi * 3 + 0];
                sely = X[bi * 3 + 1];
                selz = X[bi * 3 + 2];
            }
        }
        __syncthreads();
    }
}

// ---------------------------------------------------------------------------
// Ball query: one thread per (b,s). First NSAMPLE indices (ascending) within
// radius, padded with the first hit — matches the reference top_k trick.
// ---------------------------------------------------------------------------
__global__ void ballquery_kernel(const float* __restrict__ xyz,
                                 const float* __restrict__ newxyz,
                                 int* __restrict__ idx) {
    int t = blockIdx.x * blockDim.x + threadIdx.x;
    if (t >= NBATCH * NPOINT) return;
    int b = t / NPOINT;
    const float* X = xyz + (size_t)b * NPTS * 3;
    float qx = newxyz[(size_t)t * 3 + 0];
    float qy = newxyz[(size_t)t * 3 + 1];
    float qz = newxyz[(size_t)t * 3 + 2];
    int* out = idx + (size_t)t * NSAMPLE;
    int cnt = 0, first = 0;
    for (int n = 0; n < NPTS; ++n) {
        float dx = X[n * 3 + 0] - qx;
        float dy = X[n * 3 + 1] - qy;
        float dz = X[n * 3 + 2] - qz;
        float d2 = dx * dx + dy * dy + dz * dz;
        if (d2 <= RAD2) {
            if (cnt == 0) first = n;
            out[cnt++] = n;
            if (cnt == NSAMPLE) break;
        }
    }
    for (; cnt < NSAMPLE; ++cnt) out[cnt] = first;
}

// ---------------------------------------------------------------------------
// Fused gather + 2-layer MLP (v_wmma_f32_16x16x32_f16) + max-over-samples.
// One 8-wave workgroup per (batch, 4 query points) = 128 columns.
// ---------------------------------------------------------------------------
__global__ __launch_bounds__(256)
void mlp_kernel(const float* __restrict__ xyz,
                const float* __restrict__ feat,
                const float* __restrict__ featT, int useT,
                const float* __restrict__ newxyz,
                const int* __restrict__ idx,
                const _Float16* __restrict__ w1s,
                const _Float16* __restrict__ w2s,
                const float* __restrict__ b1v,
                const float* __restrict__ b2v,
                float* __restrict__ out) {
    __shared__ __align__(16) unsigned char smem[G_BYTES + H_BYTES];  // 60 KB
    _Float16* g    = (_Float16*)smem;              // [NCOL][G_PITCH]
    _Float16* h1   = (_Float16*)(smem + G_BYTES);  // [NCOL][H_PITCH]
    float*    pmax = (float*)smem;                 // aliases g after layer-1 barrier
    __shared__ int   sidx[NCOL];
    __shared__ float snx[4][3];

    const int tid   = threadIdx.x;
    const int b     = blockIdx.x >> 9;           // 512 tiles per batch
    const int qbase = (blockIdx.x & 511) * 4;

    // stage 0: neighbor indices + query centers for these 4 query points
    if (tid < NCOL)
        sidx[tid] = idx[((size_t)(b * NPOINT + qbase)) * NSAMPLE + tid];
    if (tid < 12)
        snx[tid / 3][tid % 3] = newxyz[((size_t)b * NPOINT + qbase) * 3 + tid];
    __syncthreads();

    // stage 1: gather into g[col][k] (f16, zero-padded K 67..95)
    for (int i = tid; i < NCOL * G_PITCH; i += 256) {
        int col = i / G_PITCH, k = i - col * G_PITCH;
        int q = col >> 5;
        int p = sidx[col];
        float v;
        if (k < 3)       v = xyz[((size_t)b * NPTS + p) * 3 + k] - snx[q][k];
        else if (k < C1) {
            int c = k - 3;
            v = useT ? featT[((size_t)b * NPTS + p) * 64 + c]
                     : feat[((size_t)b * 64 + c) * NPTS + p];
        } else v = 0.f;
        g[col * G_PITCH + k] = (_Float16)v;
    }
    __syncthreads();

    const int lane  = tid & 31;
    const int w     = tid >> 5;
    const int hi    = lane >> 4;
    const int nloc  = lane & 15;
    const int nbase = w * 16;
    const v8f zero8 = {0.f, 0.f, 0.f, 0.f, 0.f, 0.f, 0.f, 0.f};

    // ---- layer 1: h1[128 x 16cols] = relu(W1 @ g + b1) ----
    v8f acc1[8];
#pragma unroll
    for (int mt = 0; mt < 8; ++mt) acc1[mt] = zero8;
#pragma unroll
    for (int kf = 0; kf < 3; ++kf) {
        const _Float16* bp = &g[(nbase + nloc) * G_PITCH + kf * 32 + hi * 16];
        v8h blo = *(const v8h*)bp;
        v8h bhi = *(const v8h*)(bp + 8);
        v16h bf = __builtin_shufflevector(blo, bhi,
                   0,1,2,3,4,5,6,7,8,9,10,11,12,13,14,15);
#pragma unroll
        for (int mt = 0; mt < 8; ++mt) {
            v16h af = *(const v16h*)(w1s + ((size_t)(mt * 3 + kf) * 32 + lane) * 16);
            acc1[mt] = __builtin_amdgcn_wmma_f32_16x16x32_f16(
                false, af, false, bf, (short)0, acc1[mt], false, false);
        }
    }
#pragma unroll
    for (int mt = 0; mt < 8; ++mt) {
#pragma unroll
        for (int v = 0; v < 8; ++v) {
            int m = mt * 16 + v + (hi << 3);
            float x = fmaxf(acc1[mt][v] + b1v[m], 0.f);
            h1[(nbase + nloc) * H_PITCH + m] = (_Float16)x;
        }
    }
    __syncthreads();   // all reads of g complete -> pmax may alias it below

    // ---- layer 2: relu(W2 @ h1 + b2), 16 M-tiles x 4 K-frags ----
    v8f acc2[16];
#pragma unroll
    for (int mt = 0; mt < 16; ++mt) acc2[mt] = zero8;
#pragma unroll
    for (int kf = 0; kf < 4; ++kf) {
        const _Float16* bp = &h1[(nbase + nloc) * H_PITCH + kf * 32 + hi * 16];
        v8h blo = *(const v8h*)bp;
        v8h bhi = *(const v8h*)(bp + 8);
        v16h bf = __builtin_shufflevector(blo, bhi,
                   0,1,2,3,4,5,6,7,8,9,10,11,12,13,14,15);
#pragma unroll
        for (int mt = 0; mt < 16; ++mt) {
            v16h af = *(const v16h*)(w2s + ((size_t)(mt * 4 + kf) * 32 + lane) * 16);
            acc2[mt] = __builtin_amdgcn_wmma_f32_16x16x32_f16(
                false, af, false, bf, (short)0, acc2[mt], false, false);
        }
    }
    // bias + relu + row-max over this wave's 16 columns (shfl tree over n)
#pragma unroll
    for (int mt = 0; mt < 16; ++mt) {
#pragma unroll
        for (int v = 0; v < 8; ++v) {
            int m = mt * 16 + v + (hi << 3);
            float x = fmaxf(acc2[mt][v] + b2v[m], 0.f);
#pragma unroll
            for (int off = 1; off < 16; off <<= 1)
                x = fmaxf(x, __shfl_xor(x, off));
            if (nloc == 0) pmax[w * 256 + m] = x;
        }
    }
    __syncthreads();

    // combine the two waves covering each query point; store [B,256,2048]
    for (int i = tid; i < 4 * 256; i += 256) {
        int q = i >> 8, r = i & 255;
        float v = fmaxf(pmax[(2 * q) * 256 + r], pmax[(2 * q + 1) * 256 + r]);
        out[((size_t)b * H2DIM + r) * NPOINT + qbase + q] = v;
    }
}

// ---------------------------------------------------------------------------
extern "C" void kernel_launch(void* const* d_in, const int* in_sizes, int n_in,
                              void* d_out, int out_size, void* d_ws, size_t ws_size,
                              hipStream_t stream) {
    const float* xyz  = (const float*)d_in[0];
    const float* feat = (const float*)d_in[1];
    const float* W1   = (const float*)d_in[2];
    const float* b1   = (const float*)d_in[3];
    const float* W2   = (const float*)d_in[4];
    const float* b2   = (const float*)d_in[5];

    float* out     = (float*)d_out;
    float* newxyz  = out;                                   // [8,2048,3]
    float* mlpout  = out + (size_t)NBATCH * NPOINT * 3;     // [8,256,2048]

    char* ws = (char*)d_ws;
    size_t off = 0;
    int* idxbuf = (int*)(ws + off);       off += (size_t)NBATCH * NPOINT * NSAMPLE * 4; // 2 MB
    _Float16* w1s = (_Float16*)(ws + off); off += 12288 * 2;
    _Float16* w2s = (_Float16*)(ws + off); off += 32768 * 2;
    off = (off + 255) & ~(size_t)255;
    float* featT = (float*)(ws + off);
    size_t featT_bytes = (size_t)NBATCH * NPTS * 64 * 4;    // 16 MB
    int useT = (off + featT_bytes <= ws_size) ? 1 : 0;

    prep_weights<<<(12288 + 32768 + 255) / 256, 256, 0, stream>>>(W1, W2, w1s, w2s);
    if (useT)
        transpose_feat<<<NBATCH * 256, 256, 0, stream>>>(feat, featT);
    fps_kernel<<<NBATCH, 1024, 0, stream>>>(xyz, newxyz);
    ballquery_kernel<<<(NBATCH * NPOINT + 255) / 256, 256, 0, stream>>>(xyz, newxyz, idxbuf);
    mlp_kernel<<<NBATCH * (NPOINT / 4), 256, 0, stream>>>(
        xyz, feat, featT, useT, newxyz, idxbuf, w1s, w2s, b1, b2, mlpout);
    (void)in_sizes; (void)n_in; (void)out_size;
}
